// RecursiveMamba130M_43250320670932
// MI455X (gfx1250) — compile-verified, matrix operand order checked by reading, not run on hardware
//
#include <hip/hip_runtime.h>

// ---------------------------------------------------------------------------
// RecursiveMamba (768d, 2ch, 16 states, 4 loops) for MI455X / gfx1250.
//
// Math reduction: state S_i = e^{j*theta} S_{i-1} + B*x_i, y_i = Re(C*S_i)
//   =>  y_i[n] = sum_{m=0..3} G_m[n] * x_in_{i-m}[n],
//       G_m[n] = sum_s ( Cr*(cos(m*th)Br - sin(m*th)Bi)
//                       - Ci*(sin(m*th)Br + cos(m*th)Bi) ).
// So no [B,L,ch,d,16] state tensor is ever materialized; only a rolling
// 3-slot bf16 x_in history per 16-token tile, kept in LDS.
//
// Each workgroup (256 threads = 8 wave32) owns 16 tokens through all 4
// outer iterations (tokens are independent -> no global sync). Both GEMMs
// use v_wmma_f32_16x16x32_bf16 (bf16 operands / f32 accum): 8x the
// per-instruction FLOPs of the f32 16x16x4 WMMA. Weights (bf16, 4.7MB)
// stay resident in the 192MB L2.
// ---------------------------------------------------------------------------

typedef unsigned short u16;
typedef __attribute__((ext_vector_type(16))) __bf16 v16bf;
typedef __attribute__((ext_vector_type(8)))  float  v8f;

#define D_MODEL 768
#define N2D     1536      // N_CH * D_MODEL
#define TOK     16        // tokens per workgroup (= WMMA M)
#define NLOOPS  4
#define EPS_RMS 1e-6f

// LDS row pitches (elements), padded so row stride % 64 banks != 0
#define PX 776            // float pitch, x tile   (776*4  = 3104B, 16B aligned)
#define PH 776            // u16 pitch,  h tile    (776*2  = 1552B, 16B aligned)
#define PY 1544           // u16 pitch,  y / hist  (1544*2 = 3088B, 16B aligned)
#define HSLOT (TOK*PY)    // one history slot (u16 elements)

__device__ __forceinline__ u16 f2bf(float f) {
  union { float f; unsigned u; } v; v.f = f;
  unsigned r = v.u + 0x7FFFu + ((v.u >> 16) & 1u);   // round-to-nearest-even
  return (u16)(r >> 16);
}
__device__ __forceinline__ float bf2f(u16 s) {
  union { unsigned u; float f; } v; v.u = ((unsigned)s) << 16;
  return v.f;
}

// 16x32 bf16 fragment: per lane 16 bf16 = two 16B loads.
// A (MxK): lane -> M = lane&15, half = lane>>4; elems 0..7  -> K = k0+8h+e,
//          elems 8..15 -> K = k0+16+8h+(e-8).  B (KxN) mirrored with lane -> N.
union FragB16 { uint4 q[2]; v16bf v; };

// --------------------------- prep kernels ----------------------------------

__global__ void prep_win_kernel(const float* __restrict__ base,
                                const float* __restrict__ lA,
                                const float* __restrict__ lB,
                                u16* __restrict__ win_bf) {
  int idx = blockIdx.x * blockDim.x + threadIdx.x;
  if (idx >= N2D * D_MODEL) return;
  int n = idx / D_MODEL, k = idx % D_MODEL;
  float acc = base[idx];
#pragma unroll
  for (int r = 0; r < 8; ++r)
    acc += 2.0f * lB[n * 8 + r] * lA[r * D_MODEL + k];   // LORA_SCALE = 2.0
  win_bf[idx] = f2bf(acc);
}

__global__ void prep_wout_kernel(const float* __restrict__ w,
                                 u16* __restrict__ o) {
  int idx = blockIdx.x * blockDim.x + threadIdx.x;
  if (idx < D_MODEL * N2D) o[idx] = f2bf(w[idx]);
}

__global__ void prep_g_kernel(const float* __restrict__ th,
                              const float* __restrict__ br,
                              const float* __restrict__ bi,
                              const float* __restrict__ cr,
                              const float* __restrict__ ci,
                              float* __restrict__ g) {
  int idx = blockIdx.x * blockDim.x + threadIdx.x;   // flat (ch,d) in [0,1536)
  if (idx >= N2D) return;
  const int b0 = idx * 16;
#pragma unroll
  for (int m = 0; m < 4; ++m) {
    float acc = 0.f;
    for (int s = 0; s < 16; ++s) {
      float t  = th[b0 + s];
      float c  = cosf((float)m * t);
      float sn = sinf((float)m * t);
      float rr = c  * br[b0 + s] - sn * bi[b0 + s];
      float ii = sn * br[b0 + s] + c  * bi[b0 + s];
      acc += cr[b0 + s] * rr - ci[b0 + s] * ii;
    }
    g[m * N2D + idx] = acc;
  }
}

// ----------------------------- main kernel ---------------------------------

__global__ void __launch_bounds__(256)
mamba_loop_kernel(const float* __restrict__ xin,
                  const u16*   __restrict__ win_bf,   // [1536][768]
                  const u16*   __restrict__ wout_bf,  // [768][1536]
                  const float* __restrict__ gbuf,     // [4][1536]
                  const float* __restrict__ mixer_w,
                  const float* __restrict__ loop_w,
                  const float* __restrict__ step_emb, // [4][768]
                  float* __restrict__ out) {
  extern __shared__ char smem[];
  float* x_ld = (float*)smem;                                  // 16*PX f32
  u16*   h_bf = (u16*)(smem + TOK * PX * 4);                   // 16*PH bf16
  u16*   y_bf = (u16*)(smem + TOK * PX * 4 + TOK * PH * 2);    // 16*PY bf16
  u16*   hist = y_bf + TOK * PY;                               // 3 slots
  float* red1 = (float*)(hist + 3 * HSLOT);                    // [16]
  float* red2 = red1 + TOK;                                    // [16]

  const int tid  = threadIdx.x;
  const int w    = tid >> 5;        // wave id 0..7
  const int lane = tid & 31;
  const int l15  = lane & 15;
  const int half = lane >> 4;
  const int tok0 = blockIdx.x * TOK;

  // load x tile, zero history
  for (int e = tid; e < TOK * D_MODEL; e += 256) {
    int t = e / D_MODEL, j = e % D_MODEL;
    x_ld[t * PX + j] = xin[(tok0 + t) * D_MODEL + j];
  }
  for (int e = tid; e < 3 * HSLOT; e += 256) hist[e] = 0;
  __syncthreads();

  for (int it = 0; it < NLOOPS; ++it) {
    const float* emb = step_emb + it * D_MODEL;

    // h = x + emb_i (bf16 copy for GEMM1 A-operand); zero norm accumulators
    for (int e = tid; e < TOK * D_MODEL; e += 256) {
      int t = e / D_MODEL, j = e % D_MODEL;
      h_bf[t * PH + j] = f2bf(x_ld[t * PX + j] + emb[j]);
    }
    if (tid < TOK) { red1[tid] = 0.f; red2[tid] = 0.f; }
    __syncthreads();

    // ---- GEMM1: x_in = h @ W_in^T  (16 x 768 x 1536), 12 N-tiles / wave
    v8f vzero = {};
    v8f acc1[12];
#pragma unroll
    for (int q = 0; q < 12; ++q) acc1[q] = vzero;

    for (int kk = 0; kk < 24; ++kk) {
      const int k0 = kk * 32;
      FragB16 a;
      const u16* ar = h_bf + l15 * PH + k0 + 8 * half;
      a.q[0] = *(const uint4*)ar;
      a.q[1] = *(const uint4*)(ar + 16);
#pragma unroll
      for (int q = 0; q < 12; ++q) {
        const int n0 = (w * 12 + q) * 16;
        const u16* brp = win_bf + (size_t)(n0 + l15) * D_MODEL + k0 + 8 * half;
        FragB16 b;
        b.q[0] = *(const uint4*)brp;
        b.q[1] = *(const uint4*)(brp + 16);
        acc1[q] = __builtin_amdgcn_wmma_f32_16x16x32_bf16(
            false, a.v, false, b.v, (short)0, acc1[q], false, false);
      }
    }

    // ---- y_i = G0*x_in_i + G1*x_{i-1} + G2*x_{i-2} + G3*x_{i-3}
    // rolling slots: slot(k) = k mod 3; read old slot before overwrite.
    const int s1 = (it + 2) % 3, s2 = (it + 1) % 3, s3 = it % 3;
#pragma unroll
    for (int q = 0; q < 12; ++q) {
      const int n  = (w * 12 + q) * 16 + l15;     // D-frag: N = lane&15
      const float g0 = gbuf[n];
      const float g1 = gbuf[N2D + n];
      const float g2 = gbuf[2 * N2D + n];
      const float g3 = gbuf[3 * N2D + n];
#pragma unroll
      for (int r = 0; r < 8; ++r) {
        const int t = r + 8 * half;               // D-frag: M = r + 8*(lane>>4)
        const float xv = acc1[q][r];
        float yv = g0 * xv
                 + g1 * bf2f(hist[s1 * HSLOT + t * PY + n])
                 + g2 * bf2f(hist[s2 * HSLOT + t * PY + n])
                 + g3 * bf2f(hist[s3 * HSLOT + t * PY + n]);
        hist[s3 * HSLOT + t * PY + n] = f2bf(xv); // x_{i-3} dead after read
        y_bf[t * PY + n] = f2bf(yv);
      }
    }
    __syncthreads();

    // ---- GEMM2: o = y @ W_out^T  (16 x 1536 x 768), 6 J-tiles / wave
    v8f acc2[6];
#pragma unroll
    for (int q = 0; q < 6; ++q) acc2[q] = vzero;

    for (int nn = 0; nn < 48; ++nn) {
      const int n0 = nn * 32;
      FragB16 a;
      const u16* ar = y_bf + l15 * PY + n0 + 8 * half;
      a.q[0] = *(const uint4*)ar;
      a.q[1] = *(const uint4*)(ar + 16);
#pragma unroll
      for (int q = 0; q < 6; ++q) {
        const int j0 = (w * 6 + q) * 16;
        const u16* brp = wout_bf + (size_t)(j0 + l15) * N2D + n0 + 8 * half;
        FragB16 b;
        b.q[0] = *(const uint4*)brp;
        b.q[1] = *(const uint4*)(brp + 16);
        acc2[q] = __builtin_amdgcn_wmma_f32_16x16x32_bf16(
            false, a.v, false, b.v, (short)0, acc2[q], false, false);
      }
    }

    // ---- mixer RMSNorm: per-token sum of o^2 via LDS float atomics
    float p1[8] = {};
#pragma unroll
    for (int q = 0; q < 6; ++q)
#pragma unroll
      for (int r = 0; r < 8; ++r) p1[r] += acc2[q][r] * acc2[q][r];
#pragma unroll
    for (int r = 0; r < 8; ++r) atomicAdd(&red1[r + 8 * half], p1[r]);
    __syncthreads();

    float rs1[8];
#pragma unroll
    for (int r = 0; r < 8; ++r)
      rs1[r] = rsqrtf(red1[r + 8 * half] * (1.0f / D_MODEL) + EPS_RMS);

    // u = h + out' ; accumulate sum(u^2) for loop RMSNorm
    float u[6][8];
    float p2[8] = {};
#pragma unroll
    for (int q = 0; q < 6; ++q) {
      const int j = (w * 6 + q) * 16 + l15;
      const float mw = mixer_w[j];
      const float ej = emb[j];
#pragma unroll
      for (int r = 0; r < 8; ++r) {
        const int t = r + 8 * half;
        const float o  = acc2[q][r] * rs1[r] * mw;
        const float uu = x_ld[t * PX + j] + ej + o;
        u[q][r] = uu;
        p2[r] += uu * uu;
      }
    }
#pragma unroll
    for (int r = 0; r < 8; ++r) atomicAdd(&red2[r + 8 * half], p2[r]);
    __syncthreads();

    float rs2[8];
#pragma unroll
    for (int r = 0; r < 8; ++r)
      rs2[r] = rsqrtf(red2[r + 8 * half] * (1.0f / D_MODEL) + EPS_RMS);

#pragma unroll
    for (int q = 0; q < 6; ++q) {
      const int j = (w * 6 + q) * 16 + l15;
      const float lw = loop_w[j];
#pragma unroll
      for (int r = 0; r < 8; ++r) {
        const int t = r + 8 * half;
        const float xn = u[q][r] * rs2[r] * lw;
        x_ld[t * PX + j] = xn;
        if (it == NLOOPS - 1) out[(tok0 + t) * D_MODEL + j] = xn;
      }
    }
    __syncthreads();
  }
}

// ------------------------------- launcher ----------------------------------

extern "C" void kernel_launch(void* const* d_in, const int* in_sizes, int n_in,
                              void* d_out, int out_size, void* d_ws, size_t ws_size,
                              hipStream_t stream) {
  const float* x       = (const float*)d_in[0];
  const float* base    = (const float*)d_in[1];
  const float* lA      = (const float*)d_in[2];
  const float* lB      = (const float*)d_in[3];
  const float* th      = (const float*)d_in[4];
  const float* br      = (const float*)d_in[5];
  const float* bi      = (const float*)d_in[6];
  const float* cr      = (const float*)d_in[7];
  const float* ci      = (const float*)d_in[8];
  const float* wout    = (const float*)d_in[9];
  const float* mixer_w = (const float*)d_in[10];
  const float* loop_w  = (const float*)d_in[11];
  const float* emb     = (const float*)d_in[12];
  float* out = (float*)d_out;

  // workspace: bf16 W_in [1536x768], bf16 W_out [768x1536], f32 G [4x1536]
  u16*   win_bf  = (u16*)d_ws;
  u16*   wout_bf = win_bf + (size_t)N2D * D_MODEL;
  float* gbuf    = (float*)(wout_bf + (size_t)D_MODEL * N2D);

  prep_win_kernel<<<(N2D * D_MODEL + 255) / 256, 256, 0, stream>>>(base, lA, lB, win_bf);
  prep_wout_kernel<<<(D_MODEL * N2D + 255) / 256, 256, 0, stream>>>(wout, wout_bf);
  prep_g_kernel<<<(N2D + 255) / 256, 256, 0, stream>>>(th, br, bi, cr, ci, gbuf);

  const size_t lds_bytes = (size_t)TOK * PX * 4   // x tile (f32)
                         + (size_t)TOK * PH * 2   // h tile (bf16)
                         + (size_t)TOK * PY * 2   // y tile (bf16)
                         + (size_t)3 * HSLOT * 2  // x_in history (bf16)
                         + (size_t)2 * TOK * 4;   // reduction buffers
  mamba_loop_kernel<<<1024 / TOK, 256, lds_bytes, stream>>>(
      x, win_bf, wout_bf, gbuf, mixer_w, loop_w, emb, out);
}
